// Processor_23493471109945
// MI455X (gfx1250) — compile-verified, matrix operand order checked by reading, not run on hardware
//
#include <hip/hip_runtime.h>
#include <hip/hip_bf16.h>

// ---------------------------------------------------------------------------
// GNN message passing on MI455X (gfx1250), bf16 WMMA with f32 accumulation.
// Weights staged to LDS via gfx1250 async-to-LDS; A-fragments software-pipelined.
// ---------------------------------------------------------------------------

#define NN 50000
#define NE 800000
#define DD 128
#define HH 128

typedef __attribute__((ext_vector_type(16))) __bf16         v16bf;
typedef __attribute__((ext_vector_type(8)))  float          v8f;
typedef __attribute__((ext_vector_type(4)))  int            i32x4;

struct Pair16 { uint4 a, b; };

#if defined(__has_builtin)
# if __has_builtin(__builtin_amdgcn_global_load_async_to_lds_b128) && \
     __has_builtin(__builtin_amdgcn_s_wait_asynccnt)
#  define USE_ASYNC_LDS 1
# endif
#endif
#ifndef USE_ASYNC_LDS
# define USE_ASYNC_LDS 0
#endif

// Native f32 -> bf16 conversion (compiler picks v_cvt*bf16* on gfx1250).
__device__ __forceinline__ unsigned short f2bf(float f) {
  return __builtin_bit_cast(unsigned short, (__bf16)f);
}

// Cooperative copy of a contiguous global region into LDS.
// Uses GLOBAL_LOAD_ASYNC_TO_LDS_B128 (ASYNCcnt) when available.
__device__ __forceinline__ void stage_to_lds(const void* gsrc, void* ldst,
                                             int bytes, int tid, int nthr) {
#if USE_ASYNC_LDS
  auto g = (__attribute__((address_space(1))) i32x4*)(gsrc);
  auto l = (__attribute__((address_space(3))) i32x4*)(ldst);
  const int n = bytes >> 4;
  for (int i = tid; i < n; i += nthr)
    __builtin_amdgcn_global_load_async_to_lds_b128(g + i, l + i, 0, 0);
  __builtin_amdgcn_s_wait_asynccnt(0);
#else
  const uint4* g4 = (const uint4*)gsrc;
  uint4*       l4 = (uint4*)ldst;
  const int n = bytes >> 4;
  for (int i = tid; i < n; i += nthr) l4[i] = g4[i];
#endif
}

// Build a 16x32 bf16 A-fragment from an f32 row pointer.
// Lane (hi = lane>>4) holds k-locals {hi*8 .. hi*8+7} U {hi*8+16 .. hi*8+23},
// per the CDNA5 16-bit A-matrix VGPR layout.
__device__ __forceinline__ v16bf load_afrag_f32(const float* rp, int hi) {
  const int off = hi * 8;
  const float4 c0 = *(const float4*)(rp + off);
  const float4 c1 = *(const float4*)(rp + off + 4);
  const float4 c2 = *(const float4*)(rp + off + 16);
  const float4 c3 = *(const float4*)(rp + off + 20);
  v16bf t;
  t[0]  = (__bf16)c0.x; t[1]  = (__bf16)c0.y; t[2]  = (__bf16)c0.z; t[3]  = (__bf16)c0.w;
  t[4]  = (__bf16)c1.x; t[5]  = (__bf16)c1.y; t[6]  = (__bf16)c1.z; t[7]  = (__bf16)c1.w;
  t[8]  = (__bf16)c2.x; t[9]  = (__bf16)c2.y; t[10] = (__bf16)c2.z; t[11] = (__bf16)c2.w;
  t[12] = (__bf16)c3.x; t[13] = (__bf16)c3.y; t[14] = (__bf16)c3.z; t[15] = (__bf16)c3.w;
  return t;
}

// Pre-swizzled B-fragment staged in LDS: 32 contiguous bytes per lane.
__device__ __forceinline__ v16bf lds_bfrag(const unsigned short* wsrc,
                                           int kt, int nt, int lane) {
  const unsigned short* base = wsrc + (((kt * 8 + nt) * 32) + lane) * 16;
  Pair16 p;
  p.a = *(const uint4*)(base);
  p.b = *(const uint4*)(base + 8);     // contiguous 16 bf16 per lane
  return __builtin_bit_cast(v16bf, p);
}

// A-fragment from the LDS [M][col] bf16 transpose buffer (second chunk skips
// 16 k-positions per the A-matrix layout).
__device__ __forceinline__ v16bf load_afrag_lds(const unsigned short* h,
                                                int lm, int hi, int kt) {
  const unsigned short* base = h + lm * HH + kt * 32 + hi * 8;
  Pair16 p;
  p.a = *(const uint4*)(base);
  p.b = *(const uint4*)(base + 16);
  return __builtin_bit_cast(v16bf, p);
}

__device__ __forceinline__ v8f wmma_bf16(v16bf a, v16bf b, v8f c) {
  return __builtin_amdgcn_wmma_f32_16x16x32_bf16(false, a, false, b,
                                                 (short)0, c, false, false);
}

// ---------------------------------------------------------------------------
// Edge kernel: e_new = MLP(concat(x[src], x[dst], e)) + e ; agg[dst] += e_new
// 8 waves/block, one 16-edge tile per wave, weights staged in LDS.
// Dynamic LDS: 98304 (w1) + 32768 (w2) + 32768 (transpose) = 163840 B.
// ---------------------------------------------------------------------------
__global__ __launch_bounds__(256) void edge_kernel(
    const float* __restrict__ x, const long long* __restrict__ ei,
    float* __restrict__ e_buf, float* __restrict__ agg,
    const unsigned short* __restrict__ wpk,   // ew1p || ew2p contiguous
    const float* __restrict__ b1, const float* __restrict__ b2)
{
  extern __shared__ __align__(16) char smem[];
  unsigned short* w1s  = (unsigned short*)(smem);
  unsigned short* w2s  = (unsigned short*)(smem + 98304);
  unsigned short* hall = (unsigned short*)(smem + 131072);

  const int tid = threadIdx.x;
  stage_to_lds(wpk, smem, 131072, tid, 256);
  __syncthreads();

  const int wave  = tid >> 5;
  const int lane  = tid & 31;
  const int hi    = lane >> 4;
  const int lm    = lane & 15;
  const int tile  = blockIdx.x * 8 + wave;
  const int edge0 = tile * 16;

  if (edge0 < NE) {                           // wave-uniform; EXEC stays full
    const unsigned sOff = (unsigned)(int)ei[edge0 + lm] * DD;       // x[src] row
    const unsigned dOff = (unsigned)(int)ei[NE + edge0 + lm] * DD;  // x[dst] row
    const unsigned eOff = (unsigned)(edge0 + lm) * DD;              // e row
    const float* rs = x + sOff;
    const float* rd = x + dOff;
    const float* re = e_buf + eOff;

    // ---- layer 1: [16,384] @ [384,128], A software-pipelined ----
    v8f acc[8] = {};
    v16bf a_cur = load_afrag_f32(rs, hi);
#pragma unroll
    for (int kt = 0; kt < 12; ++kt) {
      v16bf a_nxt = a_cur;
      if (kt < 11) {
        const int k2 = kt + 1;
        const float* rp = (k2 < 4) ? (rs + k2 * 32)
                        : (k2 < 8) ? (rd + (k2 - 4) * 32)
                                   : (re + (k2 - 8) * 32);
        a_nxt = load_afrag_f32(rp, hi);
      }
#pragma unroll
      for (int nt = 0; nt < 8; ++nt)
        acc[nt] = wmma_bf16(a_cur, lds_bfrag(w1s, kt, nt, lane), acc[nt]);
      a_cur = a_nxt;
    }

    // bias + relu -> LDS (bf16, [M][col]) for the layer-2 A transpose
    unsigned short* h = hall + wave * (16 * HH);
#pragma unroll
    for (int nt = 0; nt < 8; ++nt) {
      const int col = nt * 16 + lm;
      const float bv = b1[col];
#pragma unroll
      for (int i = 0; i < 8; ++i) {
        float v = acc[nt][i] + bv;
        v = v > 0.f ? v : 0.f;
        h[(i + 8 * hi) * HH + col] = f2bf(v);
      }
    }

    // ---- layer 2: [16,128] @ [128,128] ----
    v8f acc2[8] = {};
#pragma unroll
    for (int kt = 0; kt < 4; ++kt) {
      const v16bf a = load_afrag_lds(h, lm, hi, kt);
#pragma unroll
      for (int nt = 0; nt < 8; ++nt)
        acc2[nt] = wmma_bf16(a, lds_bfrag(w2s, kt, nt, lane), acc2[nt]);
    }

    // ---- epilogue: residual, store e, scatter-add to agg[dst] ----
    unsigned dstOf[8];
#pragma unroll
    for (int i = 0; i < 8; ++i)
      dstOf[i] = (unsigned)(int)ei[NE + edge0 + i + 8 * hi] * DD;

#pragma unroll
    for (int nt = 0; nt < 8; ++nt) {
      const int col = nt * 16 + lm;
      const float bv = b2[col];
#pragma unroll
      for (int i = 0; i < 8; ++i) {
        const unsigned off = (unsigned)(edge0 + i + 8 * hi) * DD + col;
        const float v = acc2[nt][i] + bv + e_buf[off];
        e_buf[off] = v;
        atomicAdd(&agg[dstOf[i] + col], v);
      }
    }
  }
}

// ---------------------------------------------------------------------------
// Node kernel: x_new = MLP(concat(x, agg)) + x
// Dynamic LDS: 65536 (w1) + 32768 (w2) + 32768 (transpose) = 131072 B.
// ---------------------------------------------------------------------------
__global__ __launch_bounds__(256) void node_kernel(
    float* __restrict__ x, const float* __restrict__ agg,
    const unsigned short* __restrict__ wpk,   // nw1p || nw2p contiguous
    const float* __restrict__ b1, const float* __restrict__ b2)
{
  extern __shared__ __align__(16) char smem[];
  unsigned short* w1s  = (unsigned short*)(smem);
  unsigned short* w2s  = (unsigned short*)(smem + 65536);
  unsigned short* hall = (unsigned short*)(smem + 98304);

  const int tid = threadIdx.x;
  stage_to_lds(wpk, smem, 98304, tid, 256);
  __syncthreads();

  const int wave  = tid >> 5;
  const int lane  = tid & 31;
  const int hi    = lane >> 4;
  const int lm    = lane & 15;
  const int tile  = blockIdx.x * 8 + wave;
  const int node0 = tile * 16;

  if (node0 < NN) {
    const unsigned rOff = (unsigned)(node0 + lm) * DD;
    const float* rx = x   + rOff;
    const float* ra = agg + rOff;

    // ---- layer 1: [16,256] @ [256,128] ----
    v8f acc[8] = {};
    v16bf a_cur = load_afrag_f32(rx, hi);
#pragma unroll
    for (int kt = 0; kt < 8; ++kt) {
      v16bf a_nxt = a_cur;
      if (kt < 7) {
        const int k2 = kt + 1;
        const float* rp = (k2 < 4) ? (rx + k2 * 32) : (ra + (k2 - 4) * 32);
        a_nxt = load_afrag_f32(rp, hi);
      }
#pragma unroll
      for (int nt = 0; nt < 8; ++nt)
        acc[nt] = wmma_bf16(a_cur, lds_bfrag(w1s, kt, nt, lane), acc[nt]);
      a_cur = a_nxt;
    }

    unsigned short* h = hall + wave * (16 * HH);
#pragma unroll
    for (int nt = 0; nt < 8; ++nt) {
      const int col = nt * 16 + lm;
      const float bv = b1[col];
#pragma unroll
      for (int i = 0; i < 8; ++i) {
        float v = acc[nt][i] + bv;
        v = v > 0.f ? v : 0.f;
        h[(i + 8 * hi) * HH + col] = f2bf(v);
      }
    }

    // ---- layer 2 ----
    v8f acc2[8] = {};
#pragma unroll
    for (int kt = 0; kt < 4; ++kt) {
      const v16bf a = load_afrag_lds(h, lm, hi, kt);
#pragma unroll
      for (int nt = 0; nt < 8; ++nt)
        acc2[nt] = wmma_bf16(a, lds_bfrag(w2s, kt, nt, lane), acc2[nt]);
    }

#pragma unroll
    for (int nt = 0; nt < 8; ++nt) {
      const int col = nt * 16 + lm;
      const float bv = b2[col];
#pragma unroll
      for (int i = 0; i < 8; ++i) {
        const unsigned off = (unsigned)(node0 + i + 8 * hi) * DD + col;
        const float v = acc2[nt][i] + bv + x[off];
        x[off] = v;
      }
    }
  }
}

// ---------------------------------------------------------------------------
// Pack f32 [K,128] weights into bf16 B-fragments:
// wp[((kt*8+nt)*32 + lane)*16 + i] = bf16(w[k(kt,lane,i)*128 + n(nt,lane)])
// ---------------------------------------------------------------------------
__global__ void pack_weights(const float* __restrict__ w,
                             unsigned short* __restrict__ wp, int total) {
  const int t = blockIdx.x * 256 + threadIdx.x;
  if (t >= total) return;
  const int i    = t & 15;
  const int lane = (t >> 4) & 31;
  const int nt   = (t >> 9) & 7;
  const int kt   = t >> 12;
  const int k = kt * 32 + i + ((i >= 8) ? 8 : 0) + ((lane >= 16) ? 8 : 0);
  const int n = nt * 16 + (lane & 15);
  wp[t] = f2bf(w[k * 128 + n]);
}

__global__ void copy_f32(const float* __restrict__ s, float* __restrict__ d,
                         long long n) {
  long long t = (long long)blockIdx.x * 256 + threadIdx.x;
  const long long stride = (long long)gridDim.x * 256;
  for (; t < n; t += stride) d[t] = s[t];
}

__global__ void zero_f32(float* __restrict__ d, long long n) {
  long long t = (long long)blockIdx.x * 256 + threadIdx.x;
  const long long stride = (long long)gridDim.x * 256;
  for (; t < n; t += stride) d[t] = 0.f;
}

// ---------------------------------------------------------------------------
extern "C" void kernel_launch(void* const* d_in, const int* in_sizes, int n_in,
                              void* d_out, int out_size, void* d_ws, size_t ws_size,
                              hipStream_t stream) {
  (void)in_sizes; (void)n_in; (void)out_size; (void)ws_size;

  const float*     x_in   = (const float*)d_in[0];
  const long long* ei     = (const long long*)d_in[1];
  const float*     e_attr = (const float*)d_in[2];
  const float*     ew1    = (const float*)d_in[3];
  const float*     eb1    = (const float*)d_in[4];
  const float*     ew2    = (const float*)d_in[5];
  const float*     eb2    = (const float*)d_in[6];
  const float*     nw1    = (const float*)d_in[7];
  const float*     nb1    = (const float*)d_in[8];
  const float*     nw2    = (const float*)d_in[9];
  const float*     nb2    = (const float*)d_in[10];

  float* xbuf = (float*)d_out;                 // x lives in d_out, updated in place
  char*  ws   = (char*)d_ws;

  // workspace layout (all offsets 256B aligned; ew1p||ew2p and nw1p||nw2p contiguous)
  float*          e_buf = (float*)(ws);                       // 800000*128*4
  float*          agg   = (float*)(ws + 409600000);           // 50000*128*4
  unsigned short* ew1p  = (unsigned short*)(ws + 435200000);  // 384*128*2
  unsigned short* ew2p  = (unsigned short*)(ws + 435298304);  // 128*128*2
  unsigned short* nw1p  = (unsigned short*)(ws + 435331072);  // 256*128*2
  unsigned short* nw2p  = (unsigned short*)(ws + 435396608);  // 128*128*2

  // allow >64KB dynamic LDS (host-side config; deterministic, capture-safe)
  (void)hipFuncSetAttribute((const void*)edge_kernel,
                            hipFuncAttributeMaxDynamicSharedMemorySize, 163840);
  (void)hipFuncSetAttribute((const void*)node_kernel,
                            hipFuncAttributeMaxDynamicSharedMemorySize, 131072);

  // state init + weight packing (once per launch)
  copy_f32<<<2048, 256, 0, stream>>>(x_in, xbuf, (long long)NN * DD);
  copy_f32<<<4096, 256, 0, stream>>>(e_attr, e_buf, (long long)NE * DD);
  pack_weights<<<(384 * 128 + 255) / 256, 256, 0, stream>>>(ew1, ew1p, 384 * 128);
  pack_weights<<<(128 * 128 + 255) / 256, 256, 0, stream>>>(ew2, ew2p, 128 * 128);
  pack_weights<<<(256 * 128 + 255) / 256, 256, 0, stream>>>(nw1, nw1p, 256 * 128);
  pack_weights<<<(128 * 128 + 255) / 256, 256, 0, stream>>>(nw2, nw2p, 128 * 128);

  const int edgeBlocks = (NE / 16 + 7) / 8;    // 6250
  const int nodeBlocks = (NN / 16 + 7) / 8;    // 391

  for (int r = 0; r < 16; ++r) {
    zero_f32<<<2048, 256, 0, stream>>>(agg, (long long)NN * DD);
    edge_kernel<<<edgeBlocks, 256, 163840, stream>>>(xbuf, ei, e_buf, agg,
                                                     ew1p, eb1, eb2);
    node_kernel<<<nodeBlocks, 256, 131072, stream>>>(xbuf, agg,
                                                     nw1p, nb1, nb2);
  }
}
